// MultipleAggregationGlobalReadout_25915832664778
// MI455X (gfx1250) — matrix operand order repaired and checked by reading, not compiled, not run on hardware
//
#include <hip/hip_runtime.h>

// ---------------------------------------------------------------------------
// MI455X (gfx1250) implementation.
// Roofline: dominated by the 256 MB embedding stream (~11us @ 23.3 TB/s).
// Matmul (x_last @ W1, 4.3 GFLOP) runs on v_wmma_f32_16x16x32_f16, far below
// the memory floor. The embedding stream is loaded NON-TEMPORAL so the
// stream-once 256 MB does not rinse L2; the reused W1 fragment table (32 KB)
// and the v buffer (512 KB, re-read by the aggregate kernel) stay L2-hot.
// ---------------------------------------------------------------------------

typedef __attribute__((ext_vector_type(16))) _Float16 v16h;
typedef __attribute__((ext_vector_type(8)))  _Float16 v8h;
typedef __attribute__((ext_vector_type(8)))  float    v8f;
typedef __attribute__((ext_vector_type(4)))  float    v4f;

#define C_DIM 128
#define H_DIM 128
#define CAP   2048   // max segment size supported in LDS (E[cnt]=128, ~6 sigma = ~196)

// ---------------------------------------------------------------------------
// Kernel 1: repack W1 (C=128 x H=128, f32 row-major) into f16 WMMA B-fragments.
// Fragment layout (this code's contiguous K order, 16-bit B 32x16 per K-block): per lane
// (n = lane&15, half = lane>>4), f16 slot h in [0,16): K = kb*32 + half*16 + h.
// Flattened as [(t*4+kb)*32 + lane] * 16 + h  -> each lane loads 32 contiguous
// bytes per (t,kb) in the readout kernel.
// ---------------------------------------------------------------------------
__global__ void prep_w1_kernel(const float* __restrict__ W1,
                               _Float16* __restrict__ w1f) {
  int o = blockIdx.x * blockDim.x + threadIdx.x;
  if (o >= 8 * 4 * 32 * 16) return;
  int t    = o >> 11;        // H-tile (8 tiles of 16 columns)
  int kb   = (o >> 9) & 3;   // K-block (4 blocks of 32)
  int lane = (o >> 4) & 31;
  int h    = o & 15;
  int ncol = lane & 15;
  int hlf  = lane >> 4;
  int k    = kb * 32 + hlf * 16 + h;           // input channel c
  w1f[o] = (_Float16)W1[k * H_DIM + t * 16 + ncol];
}

// ---------------------------------------------------------------------------
// Kernel 2: node readout.  v[n] = lin(n) + relu(x_last[n] @ W1 + b1) @ W2 + b2
// Block = 128 threads = 4 waves; each wave processes 16 nodes (one M-tile).
// ---------------------------------------------------------------------------
__global__ __launch_bounds__(128)
void node_readout_kernel(const float* __restrict__ emb,
                         const float* __restrict__ W_lins,
                         const float* __restrict__ b_lins,
                         const float* __restrict__ b1,
                         const float* __restrict__ W2,
                         const float* __restrict__ b2,
                         const _Float16* __restrict__ w1f,
                         float* __restrict__ v_out,
                         int N) {
  // A-tile staging: 16 rows x 128 f16, padded to 136 to spread LDS banks.
  __shared__ __align__(16) _Float16 xl[4][16][136];
  __shared__ float slin[4][16];

  const int w    = threadIdx.x >> 5;
  const int lane = threadIdx.x & 31;
  const int hlf  = lane >> 4;
  const long node0 = ((long)blockIdx.x * 4 + w) * 16;

  // Per-lane W_lins columns (c = p*32 + lane), reused across all 16 nodes.
  float wl0[4], wl1[4], wl2[4];
#pragma unroll
  for (int p = 0; p < 4; ++p) {
    int c = p * 32 + lane;
    wl0[p] = W_lins[c];
    wl1[p] = W_lins[C_DIM + c];
    wl2[p] = W_lins[2 * C_DIM + c];
  }
  const float bl = b_lins[0] + b_lins[1] + b_lins[2];

  // ---- Stage A: stream node records (coalesced b128, NON-TEMPORAL: this is
  //      the one-shot 256 MB stream -> keep it out of L2's retained set),
  //      compute lin, stash x_last as f16 into LDS in A-fragment row layout.
  for (int m = 0; m < 16; ++m) {
    long n = node0 + m;
    float lin = 0.f;
    if (n < N) {
      const v4f* px = (const v4f*)emb + (size_t)n * C_DIM; // 128 x b128 per node
#pragma unroll
      for (int p = 0; p < 4; ++p) {
        v4f q = __builtin_nontemporal_load(&px[p * 32 + lane]); // x[n][c][0..3]
        lin = fmaf(q.x, wl0[p], lin);
        lin = fmaf(q.y, wl1[p], lin);
        lin = fmaf(q.z, wl2[p], lin);
        xl[w][m][p * 32 + lane] = (_Float16)q.w; // r = R-1 channel
      }
    } else {
#pragma unroll
      for (int p = 0; p < 4; ++p) xl[w][m][p * 32 + lane] = (_Float16)0.f;
    }
#pragma unroll
    for (int off = 16; off > 0; off >>= 1) lin += __shfl_xor(lin, off, 32);
    if (lane == 0) slin[w][m] = lin + bl;
  }
  __syncthreads();

  // ---- Stage B: WMMA. A fragments (this code's contiguous K order, 16-bit A 16x32):
  // lane row m = lane&15; slot i in [0,16) -> K = half*16 + i (matches w1f B packing).
  union AV { v16h v; v8h h2[2]; };
  AV a[4];
  const _Float16* row = &xl[w][lane & 15][0];
#pragma unroll
  for (int kb = 0; kb < 4; ++kb) {
    a[kb].h2[0] = *(const v8h*)(row + kb * 32 + hlf * 16);
    a[kb].h2[1] = *(const v8h*)(row + kb * 32 + hlf * 16 + 8);
  }

  float accnl[8];
#pragma unroll
  for (int d = 0; d < 8; ++d) accnl[d] = 0.f;

  const v16h* bf = (const v16h*)w1f;
#pragma unroll
  for (int t = 0; t < 8; ++t) {
    v8f c8 = {};
#pragma unroll
    for (int kb = 0; kb < 4; ++kb) {
      v16h bfrag = bf[(t * 4 + kb) * 32 + lane];   // 32B per lane, L2-resident
      c8 = __builtin_amdgcn_wmma_f32_16x16x32_f16(
          /*neg_a=*/false, a[kb].v, /*neg_b=*/false, bfrag,
          /*c_mod=*/(short)0, c8, /*reuse_a=*/false, /*reuse_b=*/false);
    }
    // C/D layout: VGPR d holds M = d + 8*half, N = lane&15 (column t*16+N).
    int col = t * 16 + (lane & 15);
    float b1c = b1[col];
    float w2c = W2[col];
#pragma unroll
    for (int d = 0; d < 8; ++d) {
      float hb = fmaxf(c8[d] + b1c, 0.f);
      accnl[d] = fmaf(hb, w2c, accnl[d]);
    }
  }

  // Reduce h @ W2 across the 16 lanes of each half (columns of H).
#pragma unroll
  for (int d = 0; d < 8; ++d) {
#pragma unroll
    for (int off = 1; off < 16; off <<= 1)
      accnl[d] += __shfl_xor(accnl[d], off, 32);
  }
  if ((lane & 15) == 0) {
    float b2v = b2[0];
#pragma unroll
    for (int d = 0; d < 8; ++d) {
      int m = d + 8 * hlf;
      long n = node0 + m;
      if (n < N) v_out[n] = slin[w][m] + accnl[d] + b2v;  // RT store: reused next kernel
    }
  }
}

// ---------------------------------------------------------------------------
// Kernel 3: per-batch aggregation (mean/max/min + 9 quantiles) + final MLP.
// One block per batch segment; `batch` is sorted, so bounds come from a
// binary search.  Sort in LDS with a size-adaptive bitonic network.
// ---------------------------------------------------------------------------
__global__ __launch_bounds__(256)
void aggregate_kernel(const float* __restrict__ v, const int* __restrict__ batch,
                      const float* __restrict__ Wm1, const float* __restrict__ bm1,
                      const float* __restrict__ Wm2, const float* __restrict__ bm2,
                      float* __restrict__ out, int N, int B) {
  __shared__ float sv[CAP];
  __shared__ float red[256];
  __shared__ float aggS[12];
  const int b   = blockIdx.x;
  const int tid = threadIdx.x;
  const float FMAX = 3.402823466e+38f;

  // lower_bound(batch, b) and lower_bound(batch, b+1): uniform -> scalar path.
  int lo = 0, hi = N;
  while (lo < hi) { int mid = (lo + hi) >> 1; if (batch[mid] < b) lo = mid + 1; else hi = mid; }
  int start = lo;
  lo = start; hi = N;
  while (lo < hi) { int mid = (lo + hi) >> 1; if (batch[mid] < b + 1) lo = mid + 1; else hi = mid; }
  int cnt = lo - start;
  int cl  = cnt < CAP ? cnt : CAP;

  float psum = 0.f, pmax = -FMAX, pmin = FMAX;
  for (int i = tid; i < cl; i += 256) {
    float x = v[start + i];
    sv[i] = x;
    psum += x;
    pmax = fmaxf(pmax, x);
    pmin = fminf(pmin, x);
  }
  int npow = 1;
  while (npow < cl) npow <<= 1;
  for (int i = cl + tid; i < npow; i += 256) sv[i] = FMAX;  // ascending pad
  __syncthreads();

  red[tid] = psum; __syncthreads();
  for (int s = 128; s > 0; s >>= 1) { if (tid < s) red[tid] += red[tid + s]; __syncthreads(); }
  float vsum = red[0]; __syncthreads();
  red[tid] = pmax; __syncthreads();
  for (int s = 128; s > 0; s >>= 1) { if (tid < s) red[tid] = fmaxf(red[tid], red[tid + s]); __syncthreads(); }
  float vmax = red[0]; __syncthreads();
  red[tid] = pmin; __syncthreads();
  for (int s = 128; s > 0; s >>= 1) { if (tid < s) red[tid] = fminf(red[tid], red[tid + s]); __syncthreads(); }
  float vmin = red[0]; __syncthreads();

  // Bitonic sort (ascending) over npow elements.
  for (int k = 2; k <= npow; k <<= 1) {
    for (int j = k >> 1; j > 0; j >>= 1) {
      for (int i = tid; i < npow; i += 256) {
        int ixj = i ^ j;
        if (ixj > i) {
          float x = sv[i], y = sv[ixj];
          bool up = ((i & k) == 0);
          if ((x > y) == up) { sv[i] = y; sv[ixj] = x; }
        }
      }
      __syncthreads();
    }
  }

  if (cnt > 0) {
    if (tid == 0) { aggS[0] = vsum / (float)cnt; aggS[1] = vmax; aggS[2] = vmin; }
    if (tid < 9) {
      // reference: pos = q*(cnt-1); linear interpolation on sorted values
      float pos = (float)(tid + 1) * 0.1f * (float)(cnt - 1);
      int f = (int)floorf(pos);
      float frac = pos - (float)f;
      int loi = f < cl - 1 ? f : cl - 1;
      int hii = (f + 1) < (cl - 1) ? (f + 1) : (cl - 1);
      float ql = sv[loi], qh = sv[hii];
      aggS[3 + tid] = ql + frac * (qh - ql);
    }
  } else {
    if (tid < 12) aggS[tid] = 0.f;  // jnp.where(nonempty, agg, 0)
  }
  __syncthreads();

  // out[b] = relu(agg @ Wm1 + bm1) @ Wm2 + bm2
  float partial = 0.f;
  if (tid < H_DIM) {
    float hj = bm1[tid];
#pragma unroll
    for (int i = 0; i < 12; ++i) hj = fmaf(aggS[i], Wm1[i * H_DIM + tid], hj);
    hj = fmaxf(hj, 0.f);
    partial = hj * Wm2[tid];
  }
  red[tid] = partial; __syncthreads();
  for (int s = 128; s > 0; s >>= 1) { if (tid < s) red[tid] += red[tid + s]; __syncthreads(); }
  if (tid == 0) out[b] = red[0] + bm2[0];
}

// ---------------------------------------------------------------------------
extern "C" void kernel_launch(void* const* d_in, const int* in_sizes, int n_in,
                              void* d_out, int out_size, void* d_ws, size_t ws_size,
                              hipStream_t stream) {
  const float* emb    = (const float*)d_in[0];
  const int*   batch  = (const int*)d_in[1];
  const float* W_lins = (const float*)d_in[2];
  const float* b_lins = (const float*)d_in[3];
  const float* W1     = (const float*)d_in[4];
  const float* b1     = (const float*)d_in[5];
  const float* W2     = (const float*)d_in[6];
  const float* b2     = (const float*)d_in[7];
  const float* Wm1    = (const float*)d_in[8];
  const float* bm1    = (const float*)d_in[9];
  const float* Wm2    = (const float*)d_in[10];
  const float* bm2    = (const float*)d_in[11];
  float* out = (float*)d_out;

  const int N = in_sizes[1];   // 131072 (batch index array length)
  const int B = out_size;      // 1024

  // ws layout: [0, 32KB) = W1 f16 fragments; [32KB, 32KB + 4N) = v buffer.
  _Float16* w1f = (_Float16*)d_ws;
  float* vbuf = (float*)((char*)d_ws + 32768);

  prep_w1_kernel<<<64, 256, 0, stream>>>(W1, w1f);
  int nblocks = (N + 63) / 64;  // 4 waves x 16 nodes per block
  node_readout_kernel<<<nblocks, 128, 0, stream>>>(emb, W_lins, b_lins, b1, W2, b2,
                                                   w1f, vbuf, N);
  aggregate_kernel<<<B, 256, 0, stream>>>(vbuf, batch, Wm1, bm1, Wm2, bm2, out, N, B);
}
